// AdditiveAttention_42004780155399
// MI455X (gfx1250) — matrix operand order repaired and checked by reading, not compiled, not run on hardware
//
#include <hip/hip_runtime.h>
#include <hip/hip_bf16.h>
#include <math.h>

// Problem sizes (from reference)
#define S_LEN  1024
#define B_LEN  64
#define QK_LEN 1024
#define H_LEN  1024
#define DV_LEN 1024
#define M_LEN  (S_LEN * B_LEN)      // 65536 flattened (s,b) rows
#define M_TILE 32                   // rows per workgroup (2 x 16-row subtiles)
#define KPAD   (QK_LEN + 8)         // bf16 elems per padded LDS row (2064 B: bank-conflict free)

typedef __attribute__((ext_vector_type(16))) __bf16 v16bf;
typedef __attribute__((ext_vector_type(8)))  __bf16 v8bf;
typedef __attribute__((ext_vector_type(4)))  __bf16 v4bf;
typedef __attribute__((ext_vector_type(8)))  float  v8f;
typedef __attribute__((ext_vector_type(4)))  float  v4f;

// ---------------------------------------------------------------------------
// A-fragment assembly from two 16-byte LDS loads (contiguous K order:
// lane = M%16 | (K-half<<4); element s holds K = 16*K-half + s)
// ---------------------------------------------------------------------------
__device__ inline v16bf load_a_frag(const __bf16* p) {
  v8bf lo = *(const v8bf*)p;         // K = kbase + lhi*16 .. +7
  v8bf hi = *(const v8bf*)(p + 8);   // K = kbase + lhi*16 + 8 .. +15
  return __builtin_shufflevector(lo, hi, 0,1,2,3,4,5,6,7,8,9,10,11,12,13,14,15);
}

__device__ inline v8f wmma_bf16(v16bf a, v16bf b, v8f c) {
  return __builtin_amdgcn_wmma_f32_16x16x32_bf16(
      /*neg_a=*/false, a, /*neg_b=*/false, b,
      /*c_mod=*/(short)0, c, /*reuse_a=*/false, /*reuse_b=*/false);
}

// Branch-free tanh on the hw transcendental: tanh(x) = 1 - 2/(exp(2x)+1).
// Correct limits (x->+inf: e=inf -> 1; x->-inf: e=0 -> -1); no EXEC divergence.
__device__ inline float fast_tanh(float x) {
  float e = __expf(2.0f * x);                  // v_exp_f32
  return 1.0f - __fdividef(2.0f, e + 1.0f);    // rcp-based fast divide
}

// ---------------------------------------------------------------------------
// Kernel 0: convert W_q / W_k (f32 [H,QK]) to bf16 in workspace
// ---------------------------------------------------------------------------
__global__ __launch_bounds__(256) void cvtw_kernel(
    const float* __restrict__ Wq, const float* __restrict__ Wk,
    __bf16* __restrict__ Wqb, __bf16* __restrict__ Wkb)
{
  size_t i = ((size_t)blockIdx.x * 256 + threadIdx.x) * 4;
  v4f a = *(const v4f*)(Wq + i);
  v4f b = *(const v4f*)(Wk + i);
  v4bf ab, bb;
#pragma unroll
  for (int j = 0; j < 4; ++j) { ab[j] = (__bf16)a[j]; bb[j] = (__bf16)b[j]; }
  *(v4bf*)(Wqb + i) = ab;
  *(v4bf*)(Wkb + i) = bb;
}

// ---------------------------------------------------------------------------
// Kernel 1: scores[m] = sum_h w_v[h] * tanh( A_q[m,:]·Wq[h,:] + A_k[m,:]·Wk[h,:] )
// A-stationary bf16 WMMA; both GEMMs accumulate into one f32 tile accumulator.
// ---------------------------------------------------------------------------
__global__ __launch_bounds__(256) void scores_kernel(
    const float* __restrict__ Aq_g, const float* __restrict__ Ak_g,
    const __bf16* __restrict__ Wq, const __bf16* __restrict__ Wk,
    const float* __restrict__ wv, float* __restrict__ scores)
{
  __shared__ __bf16 ldsq[M_TILE * KPAD];   // 66 KB
  __shared__ __bf16 ldsk[M_TILE * KPAD];   // 66 KB
  __shared__ float  lds_sc[M_TILE];

  const int t      = threadIdx.x;
  const int m_base = blockIdx.x * M_TILE;

  if (t < M_TILE) lds_sc[t] = 0.0f;

  // ---- Stage A tiles (f32 global -> bf16 LDS), 16B vector loads ----
  const int quads_total = M_TILE * QK_LEN / 4;       // 8192
  for (int q4 = t; q4 < quads_total; q4 += 256) {
    int row = q4 >> 8;                               // 256 quads per row
    int col = (q4 & 255) << 2;
    v4f fq = *(const v4f*)(Aq_g + (size_t)(m_base + row) * QK_LEN + col);
    v4f fk = *(const v4f*)(Ak_g + (size_t)(m_base + row) * QK_LEN + col);
    v4bf bq, bk;
#pragma unroll
    for (int j = 0; j < 4; ++j) { bq[j] = (__bf16)fq[j]; bk[j] = (__bf16)fk[j]; }
    *(v4bf*)(&ldsq[row * KPAD + col]) = bq;
    *(v4bf*)(&ldsk[row * KPAD + col]) = bk;
  }
  __syncthreads();

  const int wave = t >> 5;
  const int lane = t & 31;
  const int ln16 = lane & 15;
  const int lhi  = lane >> 4;          // 0: K-lo half, 1: K-hi half

  float p0[8], p1[8];                  // per-lane score partials (reg-resident)
#pragma unroll
  for (int i = 0; i < 8; ++i) { p0[i] = 0.0f; p1[i] = 0.0f; }

  // Each wave sweeps 8 of the 64 H-chunks; B-frags reused across 2 M-subtiles.
  for (int ni = 0; ni < 8; ++ni) {
    const int n  = wave + ni * 8;
    const int h0 = n * 16;
    v8f acc0 = {}; v8f acc1 = {};
    const __bf16* bqp = Wq + (size_t)(h0 + ln16) * QK_LEN + lhi * 16;
    const __bf16* bkp = Wk + (size_t)(h0 + ln16) * QK_LEN + lhi * 16;
    const __bf16* aq0 = &ldsq[(size_t)ln16 * KPAD + lhi * 16];
    const __bf16* ak0 = &ldsk[(size_t)ln16 * KPAD + lhi * 16];

    // Warm L0 for the *next* n-chunk's W rows (global_prefetch_b8, L2->L0).
    if (ni < 7) {
      __builtin_prefetch(bqp + 128 * QK_LEN, 0, 0);
      __builtin_prefetch(bkp + 128 * QK_LEN, 0, 0);
    }

#pragma unroll 2
    for (int k = 0; k < QK_LEN; k += 32) {
      v16bf Bq = *(const v16bf*)(bqp + k);           // 32B/lane, L2-served
      v16bf Bk = *(const v16bf*)(bkp + k);
      // Consecutive consumers of each B fragment (operand reuse, short ranges)
      acc0 = wmma_bf16(load_a_frag(aq0 + k),            Bq, acc0);
      acc1 = wmma_bf16(load_a_frag(aq0 + 16 * KPAD + k), Bq, acc1);
      acc0 = wmma_bf16(load_a_frag(ak0 + k),            Bk, acc0);
      acc1 = wmma_bf16(load_a_frag(ak0 + 16 * KPAD + k), Bk, acc1);
    }

    const float w = wv[h0 + ln16];
#pragma unroll
    for (int i = 0; i < 8; ++i) {
      p0[i] += fast_tanh(acc0[i]) * w;
      p1[i] += fast_tanh(acc1[i]) * w;
    }
  }

  // Fold partials: C/D layout is M = i + 8*lhi, N = lane%16  -> LDS atomics
#pragma unroll
  for (int i = 0; i < 8; ++i) {
    int ml = i + (lhi << 3);
    atomicAdd(&lds_sc[ml],      p0[i]);
    atomicAdd(&lds_sc[16 + ml], p1[i]);
  }
  __syncthreads();
  if (t < M_TILE) scores[m_base + t] = lds_sc[t];
}

// ---------------------------------------------------------------------------
// Kernel 2: softmax over s (per b), in place: scores -> attn
// ---------------------------------------------------------------------------
__global__ __launch_bounds__(256) void softmax_kernel(float* __restrict__ sc)
{
  __shared__ float red[256];
  const int b = blockIdx.x;
  const int t = threadIdx.x;

  float mx = -INFINITY;
  for (int s = t; s < S_LEN; s += 256) mx = fmaxf(mx, sc[s * B_LEN + b]);
  red[t] = mx; __syncthreads();
  for (int off = 128; off > 0; off >>= 1) {
    if (t < off) red[t] = fmaxf(red[t], red[t + off]);
    __syncthreads();
  }
  mx = red[0]; __syncthreads();

  float sum = 0.0f;
  for (int s = t; s < S_LEN; s += 256) sum += __expf(sc[s * B_LEN + b] - mx);
  red[t] = sum; __syncthreads();
  for (int off = 128; off > 0; off >>= 1) {
    if (t < off) red[t] += red[t + off];
    __syncthreads();
  }
  const float inv = 1.0f / red[0];

  for (int s = t; s < S_LEN; s += 256)
    sc[s * B_LEN + b] = __expf(sc[s * B_LEN + b] - mx) * inv;
}

// ---------------------------------------------------------------------------
// Kernel 3: out[b,d] = sum_s attn[s,b] * values[s,d]   (134 MFLOP, VALU)
// ---------------------------------------------------------------------------
__global__ __launch_bounds__(256) void out_kernel(
    const float* __restrict__ attn, const float* __restrict__ values,
    float* __restrict__ out)
{
  const int d = blockIdx.x * 256 + threadIdx.x;
  const int b = blockIdx.y;
  float acc = 0.0f;
#pragma unroll 4
  for (int s = 0; s < S_LEN; ++s)
    acc += attn[s * B_LEN + b] * values[(size_t)s * DV_LEN + d];
  out[(size_t)b * DV_LEN + d] = acc;
}

// ---------------------------------------------------------------------------
extern "C" void kernel_launch(void* const* d_in, const int* in_sizes, int n_in,
                              void* d_out, int out_size, void* d_ws, size_t ws_size,
                              hipStream_t stream) {
  const float* queries = (const float*)d_in[0];   // [S,B,QK]
  const float* keys    = (const float*)d_in[1];   // [S,B,QK]
  const float* values  = (const float*)d_in[2];   // [S,1,DV]
  const float* W_q     = (const float*)d_in[3];   // [H,QK]
  const float* W_k     = (const float*)d_in[4];   // [H,QK]
  const float* w_v     = (const float*)d_in[5];   // [1,H]
  float* out = (float*)d_out;                     // [B,DV]

  // Workspace layout: Wq_bf16 (2MB) | Wk_bf16 (2MB) | scores/attn (256KB)
  char* ws = (char*)d_ws;
  __bf16* Wqb   = (__bf16*)ws;
  __bf16* Wkb   = (__bf16*)(ws + (size_t)H_LEN * QK_LEN * 2);
  float*  score = (float*)(ws + (size_t)H_LEN * QK_LEN * 4);

  cvtw_kernel<<<(H_LEN * QK_LEN / 4) / 256, 256, 0, stream>>>(W_q, W_k, Wqb, Wkb);
  scores_kernel<<<M_LEN / M_TILE, 256, 0, stream>>>(queries, keys, Wqb, Wkb, w_v, score);
  softmax_kernel<<<B_LEN, 256, 0, stream>>>(score);
  out_kernel<<<dim3(DV_LEN / 256, B_LEN), 256, 0, stream>>>(score, values, out);
}
